// Heuristics_4269197492714
// MI455X (gfx1250) — compile-verified, matrix operand order checked by reading, not run on hardware
//
#include <hip/hip_runtime.h>

// CN (common-neighbors) heuristic for MI455X / gfx1250.
// out[p] = dot(adj[nodes1[p], :], adj[nodes2[p], :]),  adj entries in {0,1}.
//
// Memory-bound (0.25 FLOP/byte): 1.07 GB gathered reads -> ~46us floor at
// 23.3 TB/s. Strategy: tile 16 pairs per block, use v_wmma_f32_16x16x32_f16
// as a free 16-wide batched dot engine (diagonal of C), convert f32->{0,1}
// f16 on the fly (exact), accumulate in f32.

typedef __attribute__((ext_vector_type(16))) _Float16 v16h;
typedef __attribute__((ext_vector_type(8)))  float    v8f;
typedef __attribute__((ext_vector_type(4)))  float    f32x4;

#define N_NODES          16384
#define N_PAIRS          8192
#define PAIRS_PER_BLOCK  16
#define WAVES_PER_BLOCK  8
#define K_PER_WMMA       32
// Compile-time uniform trip count -> scalar loop branch + real unrolling.
#define K_ITERS          (N_NODES / (WAVES_PER_BLOCK * K_PER_WMMA))  // 64

__device__ __forceinline__ v16h pack16_f16(f32x4 x0, f32x4 x1, f32x4 x2, f32x4 x3) {
  // 8x v_cvt_pk_rtz_f16_f32 (RTZ is exact for {0,1} values)
  union { unsigned int u[8]; v16h h; } r;
  r.u[0] = __builtin_bit_cast(unsigned int, __builtin_amdgcn_cvt_pkrtz(x0.x, x0.y));
  r.u[1] = __builtin_bit_cast(unsigned int, __builtin_amdgcn_cvt_pkrtz(x0.z, x0.w));
  r.u[2] = __builtin_bit_cast(unsigned int, __builtin_amdgcn_cvt_pkrtz(x1.x, x1.y));
  r.u[3] = __builtin_bit_cast(unsigned int, __builtin_amdgcn_cvt_pkrtz(x1.z, x1.w));
  r.u[4] = __builtin_bit_cast(unsigned int, __builtin_amdgcn_cvt_pkrtz(x2.x, x2.y));
  r.u[5] = __builtin_bit_cast(unsigned int, __builtin_amdgcn_cvt_pkrtz(x2.z, x2.w));
  r.u[6] = __builtin_bit_cast(unsigned int, __builtin_amdgcn_cvt_pkrtz(x3.x, x3.y));
  r.u[7] = __builtin_bit_cast(unsigned int, __builtin_amdgcn_cvt_pkrtz(x3.z, x3.w));
  return r.h;
}

__global__ __launch_bounds__(256) void cn_wmma_kernel(
    const float* __restrict__ adj,
    const int*   __restrict__ nodes1,
    const int*   __restrict__ nodes2,
    float*       __restrict__ out)
{
  const int lane = threadIdx.x & 31;
  const int wave = threadIdx.x >> 5;
  const int pair_base = blockIdx.x * PAIRS_PER_BLOCK;

  const int m    = lane & 15;        // tile row (pair index within tile)
  const int koff = (lane >> 4) * 8;  // lanes 16-31 take the +8 K half

  // Per-lane gathered rows: A-lane -> rows1[m], B-lane -> rows2[m] (B fed as
  // its transpose: lane = output column n, holding rows2[n]'s K values).
  const float* __restrict__ pA =
      adj + (size_t)nodes1[pair_base + m] * N_NODES + koff;
  const float* __restrict__ pB =
      adj + (size_t)nodes2[pair_base + m] * N_NODES + koff;

  v8f c = {};  // f32 accumulator tile (16x16); diagonal = the 16 dot products

  // 8 waves split the K dimension: wave w covers kbase = w*32 + i*256.
  // Two lanes (L, L+16) of a row together consume a contiguous 128B span
  // per slice -> full-cacheline efficiency on the row gather.
  const int kstart = wave * K_PER_WMMA;
#pragma unroll 2
  for (int i = 0; i < K_ITERS; ++i) {
    const int kbase = kstart + i * (WAVES_PER_BLOCK * K_PER_WMMA);
    f32x4 a0 = *(const f32x4*)(pA + kbase);
    f32x4 a1 = *(const f32x4*)(pA + kbase + 4);
    f32x4 a2 = *(const f32x4*)(pA + kbase + 16);
    f32x4 a3 = *(const f32x4*)(pA + kbase + 20);
    f32x4 b0 = *(const f32x4*)(pB + kbase);
    f32x4 b1 = *(const f32x4*)(pB + kbase + 4);
    f32x4 b2 = *(const f32x4*)(pB + kbase + 16);
    f32x4 b3 = *(const f32x4*)(pB + kbase + 20);

    v16h a = pack16_f16(a0, a1, a2, a3);
    v16h b = pack16_f16(b0, b1, b2, b3);

    // D = A x B + C  (8 args: neg_a, A, neg_b, B, c_mod, C, reuse_a, reuse_b)
    c = __builtin_amdgcn_wmma_f32_16x16x32_f16(
        false, a, false, b, (short)0, c, false, false);
  }

  // Diagonal extraction per ISA C-layout:
  //   VGPR r, lane r      -> C[r][r]        (pair r)
  //   VGPR r, lane 24 + r -> C[8+r][8+r]    (pair 8+r)
  __shared__ float partial[WAVES_PER_BLOCK][PAIRS_PER_BLOCK];
  float val = 0.0f;
  int   pidx = -1;
#pragma unroll
  for (int r = 0; r < 8; ++r) {
    if (lane == r)      { val = c[r]; pidx = r; }
    if (lane == 24 + r) { val = c[r]; pidx = 8 + r; }
  }
  if (pidx >= 0) partial[wave][pidx] = val;
  __syncthreads();

  // Deterministic cross-wave reduction, one plain store per pair.
  if (wave == 0 && lane < PAIRS_PER_BLOCK) {
    float s = 0.0f;
#pragma unroll
    for (int w = 0; w < WAVES_PER_BLOCK; ++w) s += partial[w][lane];
    out[pair_base + lane] = s;
  }
}

extern "C" void kernel_launch(void* const* d_in, const int* in_sizes, int n_in,
                              void* d_out, int out_size, void* d_ws, size_t ws_size,
                              hipStream_t stream) {
  const float* adj    = (const float*)d_in[0];
  const int*   nodes1 = (const int*)d_in[1];
  const int*   nodes2 = (const int*)d_in[2];
  float*       out    = (float*)d_out;

  dim3 grid(N_PAIRS / PAIRS_PER_BLOCK);  // 512 blocks, disjoint 16-pair tiles
  dim3 block(256);                       // 8 wave32s per block
  cn_wmma_kernel<<<grid, block, 0, stream>>>(adj, nodes1, nodes2, out);
}
